// EdgeConvGNN_68375879352753
// MI455X (gfx1250) — matrix operand order repaired and checked by reading, not compiled, not run on hardware
//
#include <hip/hip_runtime.h>

typedef __attribute__((ext_vector_type(16))) __bf16 v16bf;
typedef __attribute__((ext_vector_type(2)))  __bf16 v2bf;
typedef __attribute__((ext_vector_type(8)))  float  v8f;

#define NGRAPH 64

__device__ __forceinline__ float frelu(float x) { return fmaxf(x, 0.0f); }

// A-fragment K index, contiguous order: k = 16*half + i
__device__ __forceinline__ int a_kidx(int i, int base8) {
    return 2 * base8 + i;
}

// pack two floats into one dword of 2x bf16 (v_cvt_pk_bf16_f32)
__device__ __forceinline__ unsigned packbf(float a, float b) {
    v2bf t;
    t[0] = (__bf16)a;
    t[1] = (__bf16)b;
    return __builtin_bit_cast(unsigned, t);
}

// view a v16bf fragment as 8 packed dwords
union frag16 {
    v16bf    v;
    unsigned u[8];
};

// order-preserving float<->uint mapping for atomicMax
__device__ __forceinline__ unsigned f2ord(float f) {
    unsigned u = __float_as_uint(f);
    return (u & 0x80000000u) ? ~u : (u | 0x80000000u);
}
__device__ __forceinline__ float ord2f(unsigned e) {
    unsigned u = (e & 0x80000000u) ? (e & 0x7FFFFFFFu) : ~e;
    return __uint_as_float(u);
}

// ---------------------------------------------------------------------------
// EdgeConv layer 1: C_in=16 -> hidden 16 -> out 16.
// Persistent waves; each iteration handles 16 edges with 2 bf16 WMMAs.
// D = W^T x msg^T: weights are the (loop-invariant) A operand.
// ---------------------------------------------------------------------------
__global__ void __launch_bounds__(256)
edgeconv1_kernel(const float* __restrict__ x,
                 const int* __restrict__ src, const int* __restrict__ dst,
                 const float* __restrict__ W1a, const float* __restrict__ b1a,
                 const float* __restrict__ W1b, const float* __restrict__ b1b,
                 float* __restrict__ hout, int E)
{
    const int  lane   = threadIdx.x & 31;
    const int  gwave  = blockIdx.x * (blockDim.x >> 5) + (threadIdx.x >> 5);
    const int  nwaves = gridDim.x * (blockDim.x >> 5);
    const int  ntile  = (E + 15) >> 4;

    const int  row   = lane & 15;              // edge within tile / N column
    const bool hi    = lane >= 16;
    const int  base8 = hi ? 8 : 0;

    // loop-invariant: A fragments (W1a^T [16x32], W1b^T padded to K=32) + biases
    v16bf aW1a, aW1b;
#pragma unroll
    for (int i = 0; i < 16; ++i) {
        int k = a_kidx(i, base8);
        aW1a[i] = (__bf16)W1a[k * 16 + row];
        aW1b[i] = (k < 16) ? (__bf16)W1b[k * 16 + row] : (__bf16)0.0f;
    }
    float bias1[8], bias2[8];
#pragma unroll
    for (int r = 0; r < 8; ++r) {
        int ch = r + (hi ? 8 : 0);
        bias1[r] = b1a[ch];
        bias2[r] = b1b[ch];
    }

    for (int tile = gwave; tile < ntile; tile += nwaves) {
        const int  eidx  = tile * 16 + row;
        const bool valid = eidx < E;
        const int  ec    = valid ? eidx : 0;
        const int  s = src[ec], d = dst[ec];

        // B fragment: msg^T [32 x 16 edges]; lanes<16 hold K=0..15 (= x_i),
        // lanes>=16 hold K=16..31 (= x_j - x_i)
        v16bf bmsg;
        {
            const float* xd = x + (size_t)d * 16;
            const float* xs = x + (size_t)s * 16;
#pragma unroll
            for (int i = 0; i < 16; ++i) {
                float xiv = xd[i];
                float v   = hi ? (xs[i] - xiv) : xiv;
                bmsg[i] = (__bf16)v;
            }
        }

        v8f zero = {};
        v8f d1 = __builtin_amdgcn_wmma_f32_16x16x32_bf16(
                     false, aW1a, false, bmsg, (short)0, zero, false, false);

        // bias + relu, then pack to bf16 pairs and shuffle packed dwords
        float h1e[8];
#pragma unroll
        for (int r = 0; r < 8; ++r) h1e[r] = frelu(d1[r] + bias1[r]);

        unsigned own[4], part[4];
#pragma unroll
        for (int p = 0; p < 4; ++p) own[p] = packbf(h1e[2 * p], h1e[2 * p + 1]);
#pragma unroll
        for (int p = 0; p < 4; ++p) part[p] = __shfl_xor(own[p], 16, 32);

        // B2: hidden [16 x 16 edges] padded to K=32 (lanes>=16 all zero).
        frag16 b2;
#pragma unroll
        for (int p = 0; p < 4; ++p) {
            b2.u[p]     = hi ? 0u : own[p];
            b2.u[p + 4] = hi ? 0u : part[p];
        }

        v8f d2 = __builtin_amdgcn_wmma_f32_16x16x32_bf16(
                     false, aW1b, false, b2.v, (short)0, zero, false, false);

        if (valid) {
            float* outp = hout + (size_t)d * 16 + (hi ? 8 : 0);
#pragma unroll
            for (int r = 0; r < 8; ++r)
                atomicAdd(outp + r, d2[r] + bias2[r]);
        }
    }
}

// ---------------------------------------------------------------------------
// EdgeConv layer 2: C_in=16 (relu'd h1) -> msg 32 -> hidden 32 -> out 32.
// 32 output channels => two M-tiles per GEMM: 4 WMMAs per 16-edge tile.
// ---------------------------------------------------------------------------
__global__ void __launch_bounds__(256)
edgeconv2_kernel(const float* __restrict__ h1,
                 const int* __restrict__ src, const int* __restrict__ dst,
                 const float* __restrict__ W2a, const float* __restrict__ b2a,
                 const float* __restrict__ W2b, const float* __restrict__ b2b,
                 float* __restrict__ hout, int E)
{
    const int  lane   = threadIdx.x & 31;
    const int  gwave  = blockIdx.x * (blockDim.x >> 5) + (threadIdx.x >> 5);
    const int  nwaves = gridDim.x * (blockDim.x >> 5);
    const int  ntile  = (E + 15) >> 4;

    const int  row   = lane & 15;
    const bool hi    = lane >= 16;
    const int  base8 = hi ? 8 : 0;

    // loop-invariant A tiles: W2a^T rows 0-15 / 16-31, W2b^T rows 0-15 / 16-31
    v16bf aAlo, aAhi, aBlo, aBhi;
#pragma unroll
    for (int i = 0; i < 16; ++i) {
        int k = a_kidx(i, base8);
        aAlo[i] = (__bf16)W2a[k * 32 + row];
        aAhi[i] = (__bf16)W2a[k * 32 + 16 + row];
        aBlo[i] = (__bf16)W2b[k * 32 + row];
        aBhi[i] = (__bf16)W2b[k * 32 + 16 + row];
    }
    float b1lo[8], b1hi[8], b2lo[8], b2hi[8];
#pragma unroll
    for (int r = 0; r < 8; ++r) {
        int ch = r + (hi ? 8 : 0);
        b1lo[r] = b2a[ch];  b1hi[r] = b2a[16 + ch];
        b2lo[r] = b2b[ch];  b2hi[r] = b2b[16 + ch];
    }

    for (int tile = gwave; tile < ntile; tile += nwaves) {
        const int  eidx  = tile * 16 + row;
        const bool valid = eidx < E;
        const int  ec    = valid ? eidx : 0;
        const int  s = src[ec], d = dst[ec];

        // B: msg^T [32 x 16 edges]; xi = relu(h1[dst]), xj = relu(h1[src])
        v16bf bmsg;
        {
            const float* xd = h1 + (size_t)d * 16;
            const float* xs = h1 + (size_t)s * 16;
#pragma unroll
            for (int i = 0; i < 16; ++i) {
                float xiv = frelu(xd[i]);
                float v   = hi ? (frelu(xs[i]) - xiv) : xiv;
                bmsg[i] = (__bf16)v;
            }
        }

        v8f zero = {};
        v8f dlo = __builtin_amdgcn_wmma_f32_16x16x32_bf16(
                      false, aAlo, false, bmsg, (short)0, zero, false, false);
        v8f dhi = __builtin_amdgcn_wmma_f32_16x16x32_bf16(
                      false, aAhi, false, bmsg, (short)0, zero, false, false);

        float hlo[8], hhi[8];
#pragma unroll
        for (int r = 0; r < 8; ++r) {
            hlo[r] = frelu(dlo[r] + b1lo[r]);
            hhi[r] = frelu(dhi[r] + b1hi[r]);
        }

        // pack to bf16 pairs, shuffle packed dwords across half-waves
        unsigned olo[4], ohi4[4], plo[4], phi[4];
#pragma unroll
        for (int p = 0; p < 4; ++p) {
            olo[p]  = packbf(hlo[2 * p], hlo[2 * p + 1]);
            ohi4[p] = packbf(hhi[2 * p], hhi[2 * p + 1]);
        }
#pragma unroll
        for (int p = 0; p < 4; ++p) {
            plo[p] = __shfl_xor(olo[p], 16, 32);
            phi[p] = __shfl_xor(ohi4[p], 16, 32);
        }

        // B2: hidden [32 x 16 edges]
        // lanes<16 hold K=0..15 (own lo + partner lo), lanes>=16 hold K=16..31
        frag16 b2;
#pragma unroll
        for (int p = 0; p < 4; ++p) {
            b2.u[p]     = hi ? phi[p]  : olo[p];
            b2.u[p + 4] = hi ? ohi4[p] : plo[p];
        }

        v8f vlo = __builtin_amdgcn_wmma_f32_16x16x32_bf16(
                      false, aBlo, false, b2.v, (short)0, zero, false, false);
        v8f vhi = __builtin_amdgcn_wmma_f32_16x16x32_bf16(
                      false, aBhi, false, b2.v, (short)0, zero, false, false);

        if (valid) {
            float* outp = hout + (size_t)d * 32 + (hi ? 8 : 0);
#pragma unroll
            for (int r = 0; r < 8; ++r) {
                atomicAdd(outp + r,      vlo[r] + b2lo[r]);
                atomicAdd(outp + 16 + r, vhi[r] + b2hi[r]);
            }
        }
    }
}

// ---------------------------------------------------------------------------
// Small epilogue kernels
// ---------------------------------------------------------------------------
__global__ void init_gmax_kernel(unsigned* gmax) {
    if (threadIdx.x < NGRAPH) gmax[threadIdx.x] = 0x007FFFFFu;  // f2ord(-inf)
}

__global__ void gate_kernel(const float* __restrict__ h2,
                            const float* __restrict__ Wg1, const float* __restrict__ bg1,
                            const float* __restrict__ Wg2, const float* __restrict__ bg2,
                            const int* __restrict__ batch,
                            float* __restrict__ gate, unsigned* __restrict__ gmax, int N)
{
    int n = blockIdx.x * blockDim.x + threadIdx.x;
    if (n >= N) return;
    float h[32];
#pragma unroll
    for (int c = 0; c < 32; ++c) h[c] = frelu(h2[(size_t)n * 32 + c]);
    float g = bg2[0];
#pragma unroll
    for (int j = 0; j < 16; ++j) {
        float t = bg1[j];
#pragma unroll
        for (int c = 0; c < 32; ++c) t += h[c] * Wg1[c * 16 + j];
        g += frelu(t) * Wg2[j];
    }
    gate[n] = g;
    atomicMax(&gmax[batch[n]], f2ord(g));
}

__global__ void expsum_kernel(const float* __restrict__ gate,
                              const int* __restrict__ batch,
                              const unsigned* __restrict__ gmax,
                              float* __restrict__ ev, float* __restrict__ denom, int N)
{
    int n = blockIdx.x * blockDim.x + threadIdx.x;
    if (n >= N) return;
    int g = batch[n];
    float e = __expf(gate[n] - ord2f(gmax[g]));
    ev[n] = e;
    atomicAdd(&denom[g], e);
}

__global__ void pool_kernel(const float* __restrict__ h2,
                            const float* __restrict__ ev, const float* __restrict__ denom,
                            const int* __restrict__ batch,
                            float* __restrict__ pooled, int N)
{
    int t = blockIdx.x * blockDim.x + threadIdx.x;
    int n = t >> 5, c = t & 31;
    if (n >= N) return;
    int g = batch[n];
    float a = ev[n] / denom[g];
    atomicAdd(&pooled[g * 32 + c], a * frelu(h2[(size_t)n * 32 + c]));
}

__global__ void cls_kernel(const float* __restrict__ pooled,
                           const float* __restrict__ Wc1, const float* __restrict__ bc1,
                           const float* __restrict__ Wc2, const float* __restrict__ bc2,
                           float* __restrict__ out)
{
    int g = threadIdx.x;
    if (g >= NGRAPH) return;
    float acc = bc2[0];
#pragma unroll
    for (int j = 0; j < 16; ++j) {
        float t = bc1[j];
#pragma unroll
        for (int c = 0; c < 32; ++c) t += pooled[g * 32 + c] * Wc1[c * 16 + j];
        acc += frelu(t) * Wc2[j];
    }
    out[g] = acc;
}

// ---------------------------------------------------------------------------
extern "C" void kernel_launch(void* const* d_in, const int* in_sizes, int n_in,
                              void* d_out, int out_size, void* d_ws, size_t ws_size,
                              hipStream_t stream)
{
    const float* x    = (const float*)d_in[0];
    const int*   eix  = (const int*)d_in[1];
    const int*   batch= (const int*)d_in[2];
    const float* W1a  = (const float*)d_in[3];  const float* b1a = (const float*)d_in[4];
    const float* W1b  = (const float*)d_in[5];  const float* b1b = (const float*)d_in[6];
    const float* W2a  = (const float*)d_in[7];  const float* b2a = (const float*)d_in[8];
    const float* W2b  = (const float*)d_in[9];  const float* b2b = (const float*)d_in[10];
    const float* Wg1  = (const float*)d_in[11]; const float* bg1 = (const float*)d_in[12];
    const float* Wg2  = (const float*)d_in[13]; const float* bg2 = (const float*)d_in[14];
    const float* Wc1  = (const float*)d_in[15]; const float* bc1 = (const float*)d_in[16];
    const float* Wc2  = (const float*)d_in[17]; const float* bc2 = (const float*)d_in[18];

    const int N = in_sizes[0] / 16;   // 100000 nodes
    const int E = in_sizes[1] / 2;    // 3200000 edges
    const int* src = eix;             // edge_index[0]
    const int* dst = eix + E;         // edge_index[1]

    char* ws = (char*)d_ws;
    float*    h1     = (float*)ws;    ws += (size_t)N * 16 * sizeof(float);
    float*    h2     = (float*)ws;    ws += (size_t)N * 32 * sizeof(float);
    float*    gate   = (float*)ws;    ws += (size_t)N * sizeof(float);
    float*    ev     = (float*)ws;    ws += (size_t)N * sizeof(float);
    unsigned* gmax   = (unsigned*)ws; ws += NGRAPH * sizeof(unsigned);
    float*    denom  = (float*)ws;    ws += NGRAPH * sizeof(float);
    float*    pooled = (float*)ws;    ws += NGRAPH * 32 * sizeof(float);

    hipMemsetAsync(h1, 0, (size_t)N * 16 * sizeof(float), stream);
    hipMemsetAsync(h2, 0, (size_t)N * 32 * sizeof(float), stream);
    hipMemsetAsync(denom, 0, NGRAPH * sizeof(float), stream);
    hipMemsetAsync(pooled, 0, NGRAPH * 32 * sizeof(float), stream);
    init_gmax_kernel<<<1, 64, 0, stream>>>(gmax);

    const int ntiles  = (E + 15) / 16;
    const int wavesPB = 8;                        // 256 threads = 8 wave32
    int blocks = (ntiles + wavesPB - 1) / wavesPB;
    if (blocks > 2048) blocks = 2048;             // persistent waves, ~12 tiles each

    edgeconv1_kernel<<<blocks, 256, 0, stream>>>(x,  src, dst, W1a, b1a, W1b, b1b, h1, E);
    edgeconv2_kernel<<<blocks, 256, 0, stream>>>(h1, src, dst, W2a, b2a, W2b, b2b, h2, E);

    gate_kernel<<<(N + 255) / 256, 256, 0, stream>>>(h2, Wg1, bg1, Wg2, bg2, batch, gate, gmax, N);
    expsum_kernel<<<(N + 255) / 256, 256, 0, stream>>>(gate, batch, gmax, ev, denom, N);
    pool_kernel<<<((N * 32) + 255) / 256, 256, 0, stream>>>(h2, ev, denom, batch, pooled, N);
    cls_kernel<<<1, 64, 0, stream>>>(pooled, Wc1, bc1, Wc2, bc2, (float*)d_out);
}
